// EquivariantBlock_61864708931786
// MI455X (gfx1250) — compile-verified, matrix operand order checked by reading, not run on hardware
//
#include <hip/hip_runtime.h>
#include <hip/hip_bf16.h>

#define NN 50000
#define EE 800000
#define HH 128
#define LL 2
#define EINF 265

constexpr float INV_NF = 0.01f;   // 1 / NF
constexpr float NORMC  = 1.0f;    // norm_constant

typedef __attribute__((ext_vector_type(16))) __bf16 v16bf;
typedef __attribute__((ext_vector_type(8)))  __bf16 v8bf;
typedef __attribute__((ext_vector_type(8)))  float  v8f;

__device__ __forceinline__ float siluf(float x) { return x / (1.0f + __expf(-x)); }
__device__ __forceinline__ float sigm(float x)  { return 1.0f / (1.0f + __expf(-x)); }
__device__ __forceinline__ void atomAddF(float* p, float v) {
  __hip_atomic_fetch_add(p, v, __ATOMIC_RELAXED, __HIP_MEMORY_SCOPE_AGENT);
}

// A fragment (16x32 bf16, ISA layout): lane holds K = base..base+7 (v0-3) and
// base+16..base+23 (v4-7); the +8 half-shift is folded into the pointer by caller.
__device__ __forceinline__ v16bf load_a_bf16(const __bf16* p) {
  v8bf lo = *(const v8bf*)(p);
  v8bf hi = *(const v8bf*)(p + 16);
  return __builtin_shufflevector(lo, hi, 0,1,2,3,4,5,6,7,8,9,10,11,12,13,14,15);
}
#define WMMA_BF16(a, b, c) \
  __builtin_amdgcn_wmma_f32_16x16x32_bf16(false, (a), false, (b), (short)0, (c), false, false)

// ---- pack a [K,128] f32 weight matrix into WMMA B-fragment order (bf16) ----
// tile(kt,nt) = 512 bf16; per lane 16 contiguous bf16 = the lane's 8 VGPRs.
__global__ void pack_w_kernel(const float* __restrict__ W, __bf16* __restrict__ dst,
                              int ktiles, long srcStride, long dstStride, int total) {
  int idx = blockIdx.x * blockDim.x + threadIdx.x;
  if (idx >= total) return;
  int e    = idx & 511;
  int tile = (idx >> 9) % (ktiles * 8);
  int mat  = idx / (512 * ktiles * 8);
  int nt = tile & 7, kt = tile >> 3;
  int laneq = e >> 4;
  int q = e & 15, v = q >> 1, h = q & 1;
  int grp = ((v >= 4) ? 2 : 0) | ((laneq >= 16) ? 1 : 0);
  int k = kt * 32 + grp * 8 + (v & 3) * 2 + h;
  int n = nt * 16 + (laneq & 15);
  dst[(size_t)mat * dstStride + (size_t)tile * 512 + e] =
      (__bf16)W[(size_t)mat * srcStride + (size_t)k * HH + n];
}

__global__ void init_x_kernel(const float* __restrict__ x, float* __restrict__ xcur,
                              __bf16* __restrict__ xb, int n) {
  int i = blockIdx.x * blockDim.x + threadIdx.x;
  if (i < n) { float v = x[i]; xcur[i] = v; xb[i] = (__bf16)v; }
}

__global__ void zero_kernel(float* __restrict__ p, int n) {
  int i = blockIdx.x * blockDim.x + threadIdx.x;
  if (i < n) p[i] = 0.0f;
}

__global__ void edge_geom_kernel(const float* __restrict__ pos, const float* __restrict__ eattr,
                                 const int* __restrict__ rows, const int* __restrict__ cols,
                                 float* __restrict__ ea, float* __restrict__ dinv) {
  int e = blockIdx.x * blockDim.x + threadIdx.x;
  if (e >= EE) return;
  int r = rows[e], c = cols[e];
  float dx = pos[r * 3 + 0] - pos[c * 3 + 0];
  float dy = pos[r * 3 + 1] - pos[c * 3 + 1];
  float dz = pos[r * 3 + 2] - pos[c * 3 + 2];
  float d2 = dx * dx + dy * dy + dz * dz;
  ea[(size_t)e * 9 + 0] = d2;
#pragma unroll
  for (int k = 0; k < 8; ++k) ea[(size_t)e * 9 + 1 + k] = eattr[(size_t)e * 8 + k];
  dinv[(size_t)e * 4 + 0] = dx;
  dinv[(size_t)e * 4 + 1] = dy;
  dinv[(size_t)e * 4 + 2] = dz;
  dinv[(size_t)e * 4 + 3] = 1.0f / (sqrtf(d2) + NORMC);
}

// ---- GCL edge MLP + attention + scatter-add (16 edges / block, 8 waves) ----
// Gathered x rows are staged once into LDS; all A-fragments come from LDS.
__global__ __launch_bounds__(256) void edge_mlp_kernel(
    const __bf16* __restrict__ xb, const float* __restrict__ ea,
    const int* __restrict__ rows, const int* __restrict__ cols,
    const __bf16* __restrict__ pW1, const float* __restrict__ W1full,
    const float* __restrict__ b1,
    const __bf16* __restrict__ pW2, const float* __restrict__ b2,
    const float* __restrict__ Watt, const float* __restrict__ batt,
    float* __restrict__ agg) {
  __shared__ int s_row[16];
  __shared__ float s_ea[16][12];
  __shared__ __attribute__((aligned(32))) __bf16 s_xr[16 * 136];
  __shared__ __attribute__((aligned(32))) __bf16 s_xc[16 * 136];
  __shared__ __attribute__((aligned(32))) __bf16 s_h1[16 * 136];
  __shared__ float s_m[16 * 132];
  __shared__ float s_gate[16];

  const int ebase = blockIdx.x * 16;
  const int tid = threadIdx.x;

  // prefetch next tile's header data (global_prefetch_b8)
  if (tid == 0 && ebase + 16 < EE) {
    __builtin_prefetch(rows + ebase + 16, 0, 3);
    __builtin_prefetch(ea + (size_t)(ebase + 16) * 9, 0, 3);
  }

  // cooperative staging: thread t stages 16B chunk c of edge i (row & col rows)
  {
    const int i = tid >> 4, c = tid & 15;
    const int rn = rows[ebase + i];
    const int cn = cols[ebase + i];
    *(v8bf*)&s_xr[i * 136 + c * 8] = *(const v8bf*)(xb + (size_t)rn * HH + c * 8);
    *(v8bf*)&s_xc[i * 136 + c * 8] = *(const v8bf*)(xb + (size_t)cn * HH + c * 8);
    if (c == 0) s_row[i] = rn;
  }
  if (tid >= 16 && tid < 160) {
    int t = tid - 16; int e = t / 9, k = t - 9 * e;
    s_ea[e][k] = ea[(size_t)(ebase + e) * 9 + k];
  }
  __syncthreads();

  const int lane = tid & 31, wid = tid >> 5;
  const int m = lane & 15, hb = lane >> 4;
  const int col  = wid * 16 + m;   // output feature this lane owns in C
  const int rowb = hb * 8;         // C rows 0-7 / 8-15

  // GEMM1: [x_row | x_col] part, K = 256 (8 bf16 WMMA k-tiles, A from LDS)
  v8f acc = {};
#pragma unroll
  for (int kt = 0; kt < 8; ++kt) {
    const __bf16* src = (kt < 4) ? s_xr : s_xc;
    v16bf a = load_a_bf16(&src[m * 136 + (kt & 3) * 32 + hb * 8]);
    v16bf b = *(const v16bf*)(pW1 + (size_t)(kt * 8 + wid) * 512 + lane * 16);
    acc = WMMA_BF16(a, b, acc);
  }
  // ea tail (K = 9, fp32) + bias + SiLU -> h1 (bf16 in LDS)
  float wk[9];
#pragma unroll
  for (int k = 0; k < 9; ++k) wk[k] = W1full[(size_t)(256 + k) * HH + col];
  const float bias1 = b1[col];
#pragma unroll
  for (int r = 0; r < 8; ++r) {
    float v = acc[r] + bias1;
#pragma unroll
    for (int k = 0; k < 9; ++k) v += s_ea[rowb + r][k] * wk[k];
    s_h1[(rowb + r) * 136 + col] = (__bf16)siluf(v);
  }
  __syncthreads();

  // GEMM2: h1 @ W2, K = 128
  v8f acc2 = {};
#pragma unroll
  for (int kt = 0; kt < 4; ++kt) {
    v16bf a = load_a_bf16(&s_h1[m * 136 + kt * 32 + hb * 8]);
    v16bf b = *(const v16bf*)(pW2 + (size_t)(kt * 8 + wid) * 512 + lane * 16);
    acc2 = WMMA_BF16(a, b, acc2);
  }
  const float bias2 = b2[col];
  float mv[8];
#pragma unroll
  for (int r = 0; r < 8; ++r) {
    float v = siluf(acc2[r] + bias2);
    mv[r] = v;
    s_m[(rowb + r) * 132 + col] = v;
  }
  __syncthreads();

  // attention gate
  if (tid < 16) {
    float dot = batt[0];
    for (int c = 0; c < HH; ++c) dot += s_m[tid * 132 + c] * Watt[c];
    s_gate[tid] = sigm(dot);
  }
  __syncthreads();

  // scatter-add (1/NF folded in)
#pragma unroll
  for (int r = 0; r < 8; ++r) {
    const int er = rowb + r;
    atomAddF(&agg[(size_t)s_row[er] * HH + col], mv[r] * s_gate[er] * INV_NF);
  }
}

// ---- node residual MLP (16 nodes / block) ----
__global__ __launch_bounds__(256) void node_mlp_kernel(
    const float* __restrict__ aggv,
    const __bf16* __restrict__ pW1, const float* __restrict__ b1,
    const __bf16* __restrict__ pW2, const float* __restrict__ b2,
    float* __restrict__ xcur, __bf16* __restrict__ xb) {
  __shared__ __attribute__((aligned(32))) __bf16 s_x[16 * 136];
  __shared__ __attribute__((aligned(32))) __bf16 s_ag[16 * 136];
  __shared__ __attribute__((aligned(32))) __bf16 s_h1[16 * 136];
  const int nbase = blockIdx.x * 16;
  const int tid = threadIdx.x;

  // staging: x rows (bf16 copy) + agg rows (f32 -> bf16 once)
  {
    const int i = tid >> 4, c = tid & 15;
    const size_t base = (size_t)(nbase + i) * HH + c * 8;
    *(v8bf*)&s_x[i * 136 + c * 8] = *(const v8bf*)(xb + base);
    v8bf g;
#pragma unroll
    for (int k = 0; k < 8; ++k) g[k] = (__bf16)aggv[base + k];
    *(v8bf*)&s_ag[i * 136 + c * 8] = g;
  }
  __syncthreads();

  const int lane = tid & 31, wid = tid >> 5;
  const int m = lane & 15, hb = lane >> 4;
  const int col = wid * 16 + m;
  const int rowb = hb * 8;

  v8f acc = {};
#pragma unroll
  for (int kt = 0; kt < 8; ++kt) {
    const __bf16* src = (kt < 4) ? s_x : s_ag;
    v16bf a = load_a_bf16(&src[m * 136 + (kt & 3) * 32 + hb * 8]);
    v16bf b = *(const v16bf*)(pW1 + (size_t)(kt * 8 + wid) * 512 + lane * 16);
    acc = WMMA_BF16(a, b, acc);
  }
  const float bias1 = b1[col];
#pragma unroll
  for (int r = 0; r < 8; ++r)
    s_h1[(rowb + r) * 136 + col] = (__bf16)siluf(acc[r] + bias1);
  __syncthreads();

  v8f acc2 = {};
#pragma unroll
  for (int kt = 0; kt < 4; ++kt) {
    v16bf a = load_a_bf16(&s_h1[m * 136 + kt * 32 + hb * 8]);
    v16bf b = *(const v16bf*)(pW2 + (size_t)(kt * 8 + wid) * 512 + lane * 16);
    acc2 = WMMA_BF16(a, b, acc2);
  }
  const float bias2 = b2[col];
#pragma unroll
  for (int r = 0; r < 8; ++r) {
    const size_t idx = (size_t)(nbase + rowb + r) * HH + col;
    float v = xcur[idx] + acc2[r] + bias2;
    xcur[idx] = v;
    xb[idx] = (__bf16)v;   // safe: all xb reads happened before first barrier
  }
}

// ---- equivariant coordinate MLP + scatter to posacc ----
__global__ __launch_bounds__(256) void coord_kernel(
    const __bf16* __restrict__ xb, const float* __restrict__ ea,
    const int* __restrict__ rows, const int* __restrict__ cols,
    const __bf16* __restrict__ pW1, const float* __restrict__ cW1full,
    const float* __restrict__ cb1,
    const __bf16* __restrict__ pW2, const float* __restrict__ cb2,
    const float* __restrict__ cW3,
    const float* __restrict__ dinv, float* __restrict__ posacc) {
  __shared__ int s_row[16];
  __shared__ float s_ea[16][12];
  __shared__ __attribute__((aligned(32))) __bf16 s_xr[16 * 136];
  __shared__ __attribute__((aligned(32))) __bf16 s_xc[16 * 136];
  __shared__ __attribute__((aligned(32))) __bf16 s_h1[16 * 136];
  __shared__ float s_m[16 * 132];

  const int ebase = blockIdx.x * 16;
  const int tid = threadIdx.x;

  if (tid == 0 && ebase + 16 < EE) {
    __builtin_prefetch(rows + ebase + 16, 0, 3);
    __builtin_prefetch(ea + (size_t)(ebase + 16) * 9, 0, 3);
  }
  {
    const int i = tid >> 4, c = tid & 15;
    const int rn = rows[ebase + i];
    const int cn = cols[ebase + i];
    *(v8bf*)&s_xr[i * 136 + c * 8] = *(const v8bf*)(xb + (size_t)rn * HH + c * 8);
    *(v8bf*)&s_xc[i * 136 + c * 8] = *(const v8bf*)(xb + (size_t)cn * HH + c * 8);
    if (c == 0) s_row[i] = rn;
  }
  if (tid >= 16 && tid < 160) {
    int t = tid - 16; int e = t / 9, k = t - 9 * e;
    s_ea[e][k] = ea[(size_t)(ebase + e) * 9 + k];
  }
  __syncthreads();

  const int lane = tid & 31, wid = tid >> 5;
  const int m = lane & 15, hb = lane >> 4;
  const int col = wid * 16 + m;
  const int rowb = hb * 8;

  v8f acc = {};
#pragma unroll
  for (int kt = 0; kt < 8; ++kt) {
    const __bf16* src = (kt < 4) ? s_xr : s_xc;
    v16bf a = load_a_bf16(&src[m * 136 + (kt & 3) * 32 + hb * 8]);
    v16bf b = *(const v16bf*)(pW1 + (size_t)(kt * 8 + wid) * 512 + lane * 16);
    acc = WMMA_BF16(a, b, acc);
  }
  float wk[9];
#pragma unroll
  for (int k = 0; k < 9; ++k) wk[k] = cW1full[(size_t)(256 + k) * HH + col];
  const float bias1 = cb1[col];
#pragma unroll
  for (int r = 0; r < 8; ++r) {
    float v = acc[r] + bias1;
#pragma unroll
    for (int k = 0; k < 9; ++k) v += s_ea[rowb + r][k] * wk[k];
    s_h1[(rowb + r) * 136 + col] = (__bf16)siluf(v);
  }
  __syncthreads();

  v8f acc2 = {};
#pragma unroll
  for (int kt = 0; kt < 4; ++kt) {
    v16bf a = load_a_bf16(&s_h1[m * 136 + kt * 32 + hb * 8]);
    v16bf b = *(const v16bf*)(pW2 + (size_t)(kt * 8 + wid) * 512 + lane * 16);
    acc2 = WMMA_BF16(a, b, acc2);
  }
  const float bias2 = cb2[col];
#pragma unroll
  for (int r = 0; r < 8; ++r)
    s_m[(rowb + r) * 132 + col] = siluf(acc2[r] + bias2);
  __syncthreads();

  if (tid < 16) {
    float phi = 0.0f;
    for (int c = 0; c < HH; ++c) phi += s_m[tid * 132 + c] * cW3[c];
    const size_t e4 = (size_t)(ebase + tid) * 4;
    const float sc = phi * dinv[e4 + 3] * INV_NF;
    const int rn = s_row[tid];
    atomAddF(&posacc[rn * 3 + 0], dinv[e4 + 0] * sc);
    atomAddF(&posacc[rn * 3 + 1], dinv[e4 + 1] * sc);
    atomAddF(&posacc[rn * 3 + 2], dinv[e4 + 2] * sc);
  }
}

__global__ void finalize_kernel(const float* __restrict__ pos, const float* __restrict__ mask,
                                const float* __restrict__ posacc, float* __restrict__ pout) {
  int i = blockIdx.x * blockDim.x + threadIdx.x;
  if (i < NN * 3) pout[i] = pos[i] + posacc[i] * mask[i / 3];
}

extern "C" void kernel_launch(void* const* d_in, const int* in_sizes, int n_in,
                              void* d_out, int out_size, void* d_ws, size_t ws_size,
                              hipStream_t stream) {
  const float* x     = (const float*)d_in[0];
  const float* pos   = (const float*)d_in[1];
  const float* mask  = (const float*)d_in[2];
  const float* eattr = (const float*)d_in[3];
  const int*   eidx  = (const int*)d_in[4];
  const float* W_e1  = (const float*)d_in[5];
  const float* b_e1  = (const float*)d_in[6];
  const float* W_e2  = (const float*)d_in[7];
  const float* b_e2  = (const float*)d_in[8];
  const float* W_att = (const float*)d_in[9];
  const float* b_att = (const float*)d_in[10];
  const float* W_n1  = (const float*)d_in[11];
  const float* b_n1  = (const float*)d_in[12];
  const float* W_n2  = (const float*)d_in[13];
  const float* b_n2  = (const float*)d_in[14];
  const float* cW1   = (const float*)d_in[15];
  const float* cb1   = (const float*)d_in[16];
  const float* cW2   = (const float*)d_in[17];
  const float* cb2   = (const float*)d_in[18];
  const float* cW3   = (const float*)d_in[19];

  float* xout = (float*)d_out;
  float* pout = xout + (size_t)NN * HH;

  char* w = (char*)d_ws;
  size_t off = 0;
  auto carve = [&](size_t bytes) -> char* {
    char* p = w + off;
    off = (off + bytes + 255) & ~(size_t)255;
    return p;
  };
  __bf16* xb     = (__bf16*)carve((size_t)NN * HH * 2);
  float*  ea     = (float*) carve((size_t)EE * 9 * 4);
  float*  dinv   = (float*) carve((size_t)EE * 4 * 4);
  float*  agg    = (float*) carve((size_t)NN * HH * 4);
  float*  posacc = (float*) carve((size_t)NN * 3 * 4);
  __bf16* pWe1 = (__bf16*)carve((size_t)LL * 8 * 8 * 512 * 2);
  __bf16* pWe2 = (__bf16*)carve((size_t)LL * 4 * 8 * 512 * 2);
  __bf16* pWn1 = (__bf16*)carve((size_t)LL * 8 * 8 * 512 * 2);
  __bf16* pWn2 = (__bf16*)carve((size_t)LL * 4 * 8 * 512 * 2);
  __bf16* pcW1 = (__bf16*)carve((size_t)8 * 8 * 512 * 2);
  __bf16* pcW2 = (__bf16*)carve((size_t)4 * 8 * 512 * 2);

  const int* rows = eidx;
  const int* cols = eidx + EE;
  auto grid1 = [](int n) { return dim3((unsigned)((n + 255) / 256)); };

  { int t = LL * 8 * 8 * 512; pack_w_kernel<<<grid1(t), 256, 0, stream>>>(W_e1, pWe1, 8, (long)EINF * HH, (long)8 * 8 * 512, t); }
  { int t = LL * 4 * 8 * 512; pack_w_kernel<<<grid1(t), 256, 0, stream>>>(W_e2, pWe2, 4, (long)HH * HH,   (long)4 * 8 * 512, t); }
  { int t = LL * 8 * 8 * 512; pack_w_kernel<<<grid1(t), 256, 0, stream>>>(W_n1, pWn1, 8, (long)2 * HH * HH, (long)8 * 8 * 512, t); }
  { int t = LL * 4 * 8 * 512; pack_w_kernel<<<grid1(t), 256, 0, stream>>>(W_n2, pWn2, 4, (long)HH * HH,   (long)4 * 8 * 512, t); }
  { int t = 8 * 8 * 512;      pack_w_kernel<<<grid1(t), 256, 0, stream>>>(cW1,  pcW1, 8, 0, 0, t); }
  { int t = 4 * 8 * 512;      pack_w_kernel<<<grid1(t), 256, 0, stream>>>(cW2,  pcW2, 4, 0, 0, t); }

  init_x_kernel<<<grid1(NN * HH), 256, 0, stream>>>(x, xout, xb, NN * HH);
  edge_geom_kernel<<<grid1(EE), 256, 0, stream>>>(pos, eattr, rows, cols, ea, dinv);

  for (int l = 0; l < LL; ++l) {
    zero_kernel<<<grid1(NN * HH), 256, 0, stream>>>(agg, NN * HH);
    edge_mlp_kernel<<<dim3(EE / 16), 256, 0, stream>>>(
        xb, ea, rows, cols,
        pWe1 + (size_t)l * 8 * 8 * 512, W_e1 + (size_t)l * EINF * HH, b_e1 + (size_t)l * HH,
        pWe2 + (size_t)l * 4 * 8 * 512, b_e2 + (size_t)l * HH,
        W_att + (size_t)l * HH, b_att + l, agg);
    node_mlp_kernel<<<dim3(NN / 16), 256, 0, stream>>>(
        agg,
        pWn1 + (size_t)l * 8 * 8 * 512, b_n1 + (size_t)l * HH,
        pWn2 + (size_t)l * 4 * 8 * 512, b_n2 + (size_t)l * HH,
        xout, xb);
  }

  zero_kernel<<<grid1(NN * 3), 256, 0, stream>>>(posacc, NN * 3);
  coord_kernel<<<dim3(EE / 16), 256, 0, stream>>>(
      xb, ea, rows, cols, pcW1, cW1, cb1, pcW2, cb2, cW3, dinv, posacc);
  finalize_kernel<<<grid1(NN * 3), 256, 0, stream>>>(pos, mask, posacc, pout);
}
